// Net_39075612459398
// MI455X (gfx1250) — matrix operand order repaired and checked by reading, not compiled, not run on hardware
//
#include <hip/hip_runtime.h>
#include <hip/hip_bf16.h>
#include <math.h>

// ---------------------------------------------------------------------------
// PointNet++ (SSG) classification forward, fp16 WMMA on gfx1250 (wave32).
//
//   1) prep_layer x9   : fold BN into (a,s); weights -> f16, pre-swizzled into
//                        per-lane WMMA B-fragment order (1 contiguous 32B/lane).
//   2) fps (2048->1024) -> c1 ; ball query r=0.2 -> nbr1/cnt1
//   3) mlp1 fused: per-center block, 64 rows, 32(pad3)->64->64->128 + masked max
//   4) fps (1024->256) -> c2 ; ball query r=0.4 -> nbr2/cnt2
//   5) mlp2 fused: gather[x1|rel] pad->160, 160->128->128->256 + masked max
//   6) mlp3: 64 rows/block, [x2|c2] pad->288, 288->256->512->1024, row-max
//   7) gmax + head (lin1/2/3 + log_softmax) -> d_out [16][10]
//
// FPS uses wave32 shfl argmax (2 barriers per sequential step).
// GEMM waves have fixed row-tile tm -> two N-tiles share each A fragment.
// ---------------------------------------------------------------------------

typedef _Float16 f16;
typedef __attribute__((ext_vector_type(16))) _Float16 v16h;
typedef __attribute__((ext_vector_type(8)))  _Float16 v8h;
typedef __attribute__((ext_vector_type(8)))  float    v8f;

#define NB_K 64   // max neighbors per center

struct LayerPtrs {
  const f16*   w;   // swizzled fragments: [Kpad/32][Cout/16][32 lanes][16 halfs]
  const float* b;   // linear bias
  const float* a;   // folded BN scale  = g * rsqrt(var+eps)
  const float* s;   // folded BN shift  = beta - a*mu
};

// ------------------------- WMMA GEMM layer helpers -------------------------
// A fragment (16x32 f16), CDNA5 lane layout:
//   lane<16 : halfs[0..7]=K(kt*32+0..7),  halfs[8..15]=K(kt*32+16..23), m=lane
//   lane>=16: halfs[0..7]=K(kt*32+8..15), halfs[8..15]=K(kt*32+24..31), m=lane-16
// B fragment (32x16 f16): same K order as A; lane n (0..15) holds K 0..7,16..23; lane n+16 holds K 8..15,24..31
// C/D (16x16 f32): vgpr r -> m = r + 8*(lane/16), n = lane%16

__device__ __forceinline__ v16h load_a_frag(const f16* actIn, int inS,
                                            int arow, int kbase) {
  v8h lo = *(const v8h*)(actIn + arow * inS + kbase);
  v8h hi = *(const v8h*)(actIn + arow * inS + kbase + 16);
  v16h a;
#pragma unroll
  for (int i = 0; i < 8; ++i) { a[i] = lo[i]; a[8 + i] = hi[i]; }
  return a;
}

// One MLP layer: LDS actIn[64][inS] (f16) -> LDS actOut[64][outS] (f16).
// z = A@W + b ; out = a*max(z,0)+s  (Linear -> ReLU -> BN(eval) folded)
// Wave w owns row-tile tm=w&3 and N-tiles tn0+2j; pairs of N-tiles share A.
__device__ __forceinline__ void layer_gemm(const f16* __restrict__ actIn, int inS,
                                           int Kpad, f16* __restrict__ actOut,
                                           int outS, int Cout, LayerPtrs lp) {
  const int lane = threadIdx.x & 31;
  const int wave = threadIdx.x >> 5;
  const int hi   = lane >> 4;
  const int l15  = lane & 15;
  const int nkt  = Kpad >> 5;
  const int nnt  = Cout >> 4;
  const int tm   = wave & 3;
  const int tn0  = wave >> 2;
  const int arow = tm * 16 + l15;
  const int jmax = nnt >> 1;  // tiles per wave (always even here)
  for (int j = 0; j < jmax; j += 2) {
    const int tnA = tn0 + 2 * j;
    const int tnB = tnA + 2;
    v8f acc0 = {}, acc1 = {};
    for (int kt = 0; kt < nkt; ++kt) {
      v16h af  = load_a_frag(actIn, inS, arow, kt * 32 + hi * 8);
      v16h bf0 = *(const v16h*)(lp.w + (((size_t)kt * nnt + tnA) * 32 + lane) * 16);
      v16h bf1 = *(const v16h*)(lp.w + (((size_t)kt * nnt + tnB) * 32 + lane) * 16);
      acc0 = __builtin_amdgcn_wmma_f32_16x16x32_f16(false, af, false, bf0,
                                                    (short)0, acc0, false, false);
      acc1 = __builtin_amdgcn_wmma_f32_16x16x32_f16(false, af, false, bf1,
                                                    (short)0, acc1, false, false);
    }
#pragma unroll
    for (int q = 0; q < 2; ++q) {
      const v8f acc = q ? acc1 : acc0;
      const int col = (q ? tnB : tnA) * 16 + l15;
      const float bb = lp.b[col], av = lp.a[col], sv = lp.s[col];
#pragma unroll
      for (int r = 0; r < 8; ++r) {
        const int m = tm * 16 + hi * 8 + r;
        float z = av * fmaxf(acc[r] + bb, 0.0f) + sv;
        actOut[m * outS + col] = (f16)z;
      }
    }
  }
}

// Final layer of an SA block: GEMM then max over valid rows (row m valid iff
// m < cnt). Per-tile row-max reduced in-register + shfl, partials to red[4][Cout].
__device__ __forceinline__ void layer_gemm_max(const f16* __restrict__ actIn,
                                               int inS, int Kpad, int Cout,
                                               LayerPtrs lp, int cnt,
                                               float* __restrict__ red) {
  const int lane = threadIdx.x & 31;
  const int wave = threadIdx.x >> 5;
  const int hi   = lane >> 4;
  const int l15  = lane & 15;
  const int nkt  = Kpad >> 5;
  const int nnt  = Cout >> 4;
  const int tm   = wave & 3;
  const int tn0  = wave >> 2;
  const int arow = tm * 16 + l15;
  const int jmax = nnt >> 1;
  for (int j = 0; j < jmax; j += 2) {
    const int tnA = tn0 + 2 * j;
    const int tnB = tnA + 2;
    v8f acc0 = {}, acc1 = {};
    for (int kt = 0; kt < nkt; ++kt) {
      v16h af  = load_a_frag(actIn, inS, arow, kt * 32 + hi * 8);
      v16h bf0 = *(const v16h*)(lp.w + (((size_t)kt * nnt + tnA) * 32 + lane) * 16);
      v16h bf1 = *(const v16h*)(lp.w + (((size_t)kt * nnt + tnB) * 32 + lane) * 16);
      acc0 = __builtin_amdgcn_wmma_f32_16x16x32_f16(false, af, false, bf0,
                                                    (short)0, acc0, false, false);
      acc1 = __builtin_amdgcn_wmma_f32_16x16x32_f16(false, af, false, bf1,
                                                    (short)0, acc1, false, false);
    }
#pragma unroll
    for (int q = 0; q < 2; ++q) {
      const v8f acc = q ? acc1 : acc0;
      const int col = (q ? tnB : tnA) * 16 + l15;
      const float bb = lp.b[col], av = lp.a[col], sv = lp.s[col];
      float vmax = -INFINITY;
#pragma unroll
      for (int r = 0; r < 8; ++r) {
        const int m = tm * 16 + hi * 8 + r;
        float z = av * fmaxf(acc[r] + bb, 0.0f) + sv;
        vmax = (m < cnt) ? fmaxf(vmax, z) : vmax;
      }
      vmax = fmaxf(vmax, __shfl_xor(vmax, 16, 32));  // rows m and m+8
      if (hi == 0) red[tm * Cout + col] = vmax;
    }
  }
}

// ----------------------------- weight prep ---------------------------------
__global__ void prep_layer_kernel(const float* __restrict__ w,
                                  const float* __restrict__ b,
                                  const float* __restrict__ g,
                                  const float* __restrict__ beta,
                                  const float* __restrict__ mu,
                                  const float* __restrict__ var,
                                  int Cin, int Cout, int Kpad,
                                  f16* __restrict__ wout, float* __restrict__ bo,
                                  float* __restrict__ ao, float* __restrict__ so) {
  const int tid = blockIdx.x * 256 + threadIdx.x;
  const int total = Kpad * Cout;
  if (tid < total) {
    const int h    = tid & 15;
    const int lane = (tid >> 4) & 31;
    const int frag = tid >> 9;
    const int nnt  = Cout >> 4;
    const int tn   = frag % nnt;
    const int kt   = frag / nnt;
    const int n = tn * 16 + (lane & 15);
    const int k = kt * 32 + ((lane >> 4) << 3) + (h & 7) + ((h >> 3) << 4);
    const float v = (k < Cin) ? w[(size_t)k * Cout + n] : 0.0f;
    wout[tid] = (f16)v;
  }
  if (tid < Cout) {
    const float a = g[tid] * rsqrtf(var[tid] + 1e-5f);
    bo[tid] = b[tid];
    ao[tid] = a;
    so[tid] = beta[tid] - a * mu[tid];
  }
}

// ------------------------- furthest point sampling -------------------------
// Sequential over S-1 steps; per step: register-resident min-dist update,
// wave32 shfl argmax (tie -> lowest index), 8 wave results via LDS, every
// thread reduces them redundantly so `last` is uniform. 2 barriers per step.
template <int T, int S, int PT>  // PT = T/256
__global__ __launch_bounds__(256) void fps_kernel(const float* __restrict__ pos,
                                                  int* __restrict__ idx,
                                                  float* __restrict__ ctr) {
  const int cloud = blockIdx.x;
  const int tid   = threadIdx.x;
  const int lane  = tid & 31;
  const int wave  = tid >> 5;
  const float* p  = pos + (size_t)cloud * T * 3;
  __shared__ float rv[8];
  __shared__ int   ri[8];
  float px[PT], py[PT], pz[PT], mind[PT];
#pragma unroll
  for (int i = 0; i < PT; ++i) {
    const int t = tid + i * 256;
    px[i] = p[t * 3 + 0]; py[i] = p[t * 3 + 1]; pz[i] = p[t * 3 + 2];
    mind[i] = INFINITY;
  }
  if (tid == 0) {
    idx[cloud * S] = 0;
    ctr[(size_t)cloud * S * 3 + 0] = p[0];
    ctr[(size_t)cloud * S * 3 + 1] = p[1];
    ctr[(size_t)cloud * S * 3 + 2] = p[2];
  }
  int last = 0;
  for (int s = 1; s < S; ++s) {
    const float lx = p[last * 3 + 0], ly = p[last * 3 + 1], lz = p[last * 3 + 2];
    float bv = -1.0f;
    int   bi = T;
#pragma unroll
    for (int i = 0; i < PT; ++i) {
      const float dx = px[i] - lx, dy = py[i] - ly, dz = pz[i] - lz;
      const float d = dx * dx + dy * dy + dz * dz;
      mind[i] = fminf(mind[i], d);
      if (mind[i] > bv) { bv = mind[i]; bi = tid + i * 256; }
    }
    // wave32 argmax, first-occurrence tie-break
#pragma unroll
    for (int off = 16; off; off >>= 1) {
      const float v2 = __shfl_xor(bv, off, 32);
      const int   i2 = __shfl_xor(bi, off, 32);
      if (v2 > bv || (v2 == bv && i2 < bi)) { bv = v2; bi = i2; }
    }
    if (lane == 0) { rv[wave] = bv; ri[wave] = bi; }
    __syncthreads();
    float cv = rv[0];
    int   ci = ri[0];
#pragma unroll
    for (int w = 1; w < 8; ++w) {
      const float v2 = rv[w];
      const int   i2 = ri[w];
      if (v2 > cv || (v2 == cv && i2 < ci)) { cv = v2; ci = i2; }
    }
    last = ci;  // uniform across block
    if (tid == 0) {
      idx[cloud * S + s] = last;
      ctr[((size_t)cloud * S + s) * 3 + 0] = p[last * 3 + 0];
      ctr[((size_t)cloud * S + s) * 3 + 1] = p[last * 3 + 1];
      ctr[((size_t)cloud * S + s) * 3 + 2] = p[last * 3 + 2];
    }
    __syncthreads();  // protect rv/ri for next step
  }
}

// ------------------------------ ball query ---------------------------------
// First-K in-ball points by ascending index (matches top_k-of-index semantics;
// valid neighbors form a prefix of length cnt).
__global__ void ballq_kernel(const float* __restrict__ pts,
                             const float* __restrict__ ctr,
                             int* __restrict__ nbr, int* __restrict__ cnt,
                             int T, int S, float r2) {
  const int gid = blockIdx.x * 256 + threadIdx.x;
  const int cloud = gid / S, s = gid % S;
  const float* p = pts + (size_t)cloud * T * 3;
  const float cx = ctr[((size_t)cloud * S + s) * 3 + 0];
  const float cy = ctr[((size_t)cloud * S + s) * 3 + 1];
  const float cz = ctr[((size_t)cloud * S + s) * 3 + 2];
  int* out = nbr + ((size_t)cloud * S + s) * NB_K;
  int c = 0;
  for (int j = 0; j < T && c < NB_K; ++j) {
    const float dx = cx - p[j * 3 + 0];
    const float dy = cy - p[j * 3 + 1];
    const float dz = cz - p[j * 3 + 2];
    if (dx * dx + dy * dy + dz * dz <= r2) out[c++] = j;
  }
  cnt[(size_t)cloud * S + s] = c;
  for (int k = c; k < NB_K; ++k) out[k] = 0;
}

// ------------------------------ SA1: mlp1 ----------------------------------
// One block per center: 64 neighbor rows, channels 32(pad3)->64->64->128.
#define MLP1_LDS (64 * 40 * 2 + 64 * 72 * 2 + 64 * 72 * 2 + 4 * 128 * 4)
__global__ __launch_bounds__(256) void mlp1_kernel(
    const float* __restrict__ pos, const float* __restrict__ c1,
    const int* __restrict__ nbr1, const int* __restrict__ cnt1,
    LayerPtrs l0, LayerPtrs l1, LayerPtrs l2, f16* __restrict__ x1) {
  extern __shared__ char smem[];
  f16* X0 = (f16*)smem;                      // [64][40]
  f16* X1 = X0 + 64 * 40;                    // [64][72]
  f16* X2 = X1 + 64 * 72;                    // [64][72]
  float* red = (float*)(X2 + 64 * 72);       // [4][128]
  const int blk = blockIdx.x;
  const int cloud = blk >> 10, s = blk & 1023;
  const int base = cloud * 1024 + s;
  const int cnt = cnt1[base];
  const int tid = threadIdx.x;
  if (tid < 64) {
    const int j = nbr1[(size_t)base * NB_K + tid];
    const float cx = c1[(size_t)base * 3 + 0];
    const float cy = c1[(size_t)base * 3 + 1];
    const float cz = c1[(size_t)base * 3 + 2];
    const float* pp = pos + ((size_t)cloud * 2048 + j) * 3;
    f16* row = X0 + tid * 40;
    row[0] = (f16)(pp[0] - cx);
    row[1] = (f16)(pp[1] - cy);
    row[2] = (f16)(pp[2] - cz);
    for (int c = 3; c < 32; ++c) row[c] = (f16)0.0f;
  }
  __syncthreads();
  layer_gemm(X0, 40, 32, X1, 72, 64, l0);
  __syncthreads();
  layer_gemm(X1, 72, 64, X2, 72, 64, l1);
  __syncthreads();
  layer_gemm_max(X2, 72, 64, 128, l2, cnt, red);
  __syncthreads();
  for (int ch = tid; ch < 128; ch += 256) {
    const float m = fmaxf(fmaxf(red[ch], red[128 + ch]),
                          fmaxf(red[256 + ch], red[384 + ch]));
    x1[(size_t)base * 128 + ch] = (f16)m;
  }
}

// ------------------------------ SA2: mlp2 ----------------------------------
// One block per center: 64 rows, [x1(128)|rel(3)] pad->160 ->128->128->256.
#define MLP2_LDS (64 * 168 * 2 + 64 * 136 * 2 + 64 * 136 * 2 + 4 * 256 * 4 + 64 * 4)
__global__ __launch_bounds__(256) void mlp2_kernel(
    const f16* __restrict__ x1, const float* __restrict__ c1,
    const float* __restrict__ c2, const int* __restrict__ nbr2,
    const int* __restrict__ cnt2, LayerPtrs l0, LayerPtrs l1, LayerPtrs l2,
    f16* __restrict__ x2) {
  extern __shared__ char smem[];
  f16* X0 = (f16*)smem;                      // [64][168]
  f16* X1 = X0 + 64 * 168;                   // [64][136]
  f16* X2 = X1 + 64 * 136;                   // [64][136]
  float* red = (float*)(X2 + 64 * 136);      // [4][256]
  int* nid = (int*)(red + 4 * 256);          // [64]
  const int blk = blockIdx.x;
  const int cloud = blk >> 8, s = blk & 255;
  const int base = cloud * 256 + s;
  const int cnt = cnt2[base];
  const int tid = threadIdx.x;
  if (tid < 64) nid[tid] = nbr2[(size_t)base * NB_K + tid];
  __syncthreads();
  const float cx = c2[(size_t)base * 3 + 0];
  const float cy = c2[(size_t)base * 3 + 1];
  const float cz = c2[(size_t)base * 3 + 2];
  for (int idx = tid; idx < 64 * 160; idx += 256) {
    const int row = idx / 160, col = idx % 160;
    const int j = nid[row];
    f16 v;
    if (col < 128) {
      v = x1[((size_t)cloud * 1024 + j) * 128 + col];
    } else if (col < 131) {
      const float pv = c1[((size_t)cloud * 1024 + j) * 3 + (col - 128)];
      const float cc = (col == 128) ? cx : ((col == 129) ? cy : cz);
      v = (f16)(pv - cc);
    } else {
      v = (f16)0.0f;
    }
    X0[row * 168 + col] = v;
  }
  __syncthreads();
  layer_gemm(X0, 168, 160, X1, 136, 128, l0);
  __syncthreads();
  layer_gemm(X1, 136, 128, X2, 136, 128, l1);
  __syncthreads();
  layer_gemm_max(X2, 136, 128, 256, l2, cnt, red);
  __syncthreads();
  for (int ch = tid; ch < 256; ch += 256) {
    const float m = fmaxf(fmaxf(red[ch], red[256 + ch]),
                          fmaxf(red[512 + ch], red[768 + ch]));
    x2[(size_t)base * 256 + ch] = (f16)m;
  }
}

// -------------------------------- mlp3 -------------------------------------
// 4 blocks/cloud, 64 rows each: [x2(256)|c2(3)] pad->288 ->256->512->1024,
// row-max -> partials[cloud][rb][1024].
#define MLP3_BIG_OFF (64 * 264 * 2)                   // X1 region size (33792)
#define MLP3_RED_OFF (MLP3_BIG_OFF + 64 * 520 * 2)    // + BIG (66560) = 100352
#define MLP3_LDS     (MLP3_RED_OFF + 4 * 1024 * 4)    // + red (16384) = 116736
__global__ __launch_bounds__(256) void mlp3_kernel(
    const f16* __restrict__ x2, const float* __restrict__ c2,
    LayerPtrs l0, LayerPtrs l1, LayerPtrs l2, float* __restrict__ part) {
  extern __shared__ char smem[];
  f16* X1 = (f16*)smem;                          // [64][264]
  f16* X0 = (f16*)(smem + MLP3_BIG_OFF);         // [64][296] (overlaps X2; X0
  f16* X2b = (f16*)(smem + MLP3_BIG_OFF);        //  is dead before X2 written)
  float* red = (float*)(smem + MLP3_RED_OFF);    // [4][1024]
  const int blk = blockIdx.x;
  const int cloud = blk >> 2, rb = blk & 3;
  const int tid = threadIdx.x;
  for (int idx = tid; idx < 64 * 288; idx += 256) {
    const int row = idx / 288, col = idx % 288;
    const int s = rb * 64 + row;
    f16 v;
    if (col < 256)       v = x2[((size_t)cloud * 256 + s) * 256 + col];
    else if (col < 259)  v = (f16)c2[((size_t)cloud * 256 + s) * 3 + (col - 256)];
    else                 v = (f16)0.0f;
    X0[row * 296 + col] = v;
  }
  __syncthreads();
  layer_gemm(X0, 296, 288, X1, 264, 256, l0);
  __syncthreads();
  layer_gemm(X1, 264, 256, X2b, 520, 512, l1);   // overwrites X0 (dead)
  __syncthreads();
  layer_gemm_max(X2b, 520, 512, 1024, l2, 64, red);
  __syncthreads();
  for (int ch = tid; ch < 1024; ch += 256) {
    const float m = fmaxf(fmaxf(red[ch], red[1024 + ch]),
                          fmaxf(red[2048 + ch], red[3072 + ch]));
    part[((size_t)cloud * 4 + rb) * 1024 + ch] = m;
  }
}

__global__ void gmax_kernel(const float* __restrict__ part, float* __restrict__ g) {
  const int i = blockIdx.x * 256 + threadIdx.x;  // 16*1024 total
  const int cloud = i >> 10, ch = i & 1023;
  const float* p = part + (size_t)cloud * 4096 + ch;
  g[i] = fmaxf(fmaxf(p[0], p[1024]), fmaxf(p[2048], p[3072]));
}

// ------------------------------ classifier ---------------------------------
__global__ __launch_bounds__(256) void head_kernel(
    const float* __restrict__ g, const float* __restrict__ w1,
    const float* __restrict__ b1, const float* __restrict__ w2,
    const float* __restrict__ b2, const float* __restrict__ w3,
    const float* __restrict__ b3, float* __restrict__ out) {
  __shared__ float h1[512], h2[256], logits[10];
  const int cloud = blockIdx.x, tid = threadIdx.x;
  const float* gv = g + (size_t)cloud * 1024;
  for (int o = tid; o < 512; o += 256) {
    float acc = b1[o];
    for (int k = 0; k < 1024; ++k) acc += gv[k] * w1[(size_t)k * 512 + o];
    h1[o] = fmaxf(acc, 0.0f);
  }
  __syncthreads();
  for (int o = tid; o < 256; o += 256) {
    float acc = b2[o];
    for (int k = 0; k < 512; ++k) acc += h1[k] * w2[(size_t)k * 256 + o];
    h2[o] = fmaxf(acc, 0.0f);
  }
  __syncthreads();
  if (tid < 10) {
    float acc = b3[tid];
    for (int k = 0; k < 256; ++k) acc += h2[k] * w3[(size_t)k * 10 + tid];
    logits[tid] = acc;
  }
  __syncthreads();
  if (tid == 0) {
    float mx = logits[0];
    for (int i = 1; i < 10; ++i) mx = fmaxf(mx, logits[i]);
    float ssum = 0.0f;
    for (int i = 0; i < 10; ++i) ssum += expf(logits[i] - mx);
    const float lse = mx + logf(ssum);
    for (int i = 0; i < 10; ++i) out[cloud * 10 + i] = logits[i] - lse;
  }
}

// ------------------------------- launcher ----------------------------------
extern "C" void kernel_launch(void* const* d_in, const int* in_sizes, int n_in,
                              void* d_out, int out_size, void* d_ws, size_t ws_size,
                              hipStream_t stream) {
  (void)in_sizes; (void)n_in; (void)out_size; (void)ws_size;
  // Input flattening assumption (insertion order of setup_inputs()):
  //  [0]=pos[16,2048,3]; then mlp1..mlp3, each 3 layers of (w,b,g,beta,mu,var);
  //  then lin1 w,b ; lin2 w,b ; lin3 w,b  (61 arrays total).
  const float* POS = (const float*)d_in[0];
  auto LIN = [&](int mlp, int layer, int item) -> const float* {
    return (const float*)d_in[1 + mlp * 18 + layer * 6 + item];
  };
  const float* W1 = (const float*)d_in[55]; const float* B1 = (const float*)d_in[56];
  const float* W2 = (const float*)d_in[57]; const float* B2 = (const float*)d_in[58];
  const float* W3 = (const float*)d_in[59]; const float* B3 = (const float*)d_in[60];

  static const int CIN[9]  = {3, 64, 64, 131, 128, 128, 259, 256, 512};
  static const int COUT[9] = {64, 64, 128, 128, 128, 256, 256, 512, 1024};
  static const int KPAD[9] = {32, 64, 64, 160, 128, 128, 288, 256, 512};

  char* ws = (char*)d_ws;
  size_t off = 0;
  auto take = [&](size_t bytes) -> void* {
    void* p = ws + off;
    off += (bytes + 255) & ~(size_t)255;
    return p;
  };
  int*   idx1 = (int*)take(16 * 1024 * 4);
  float* c1   = (float*)take(16 * 1024 * 3 * 4);
  int*   nbr1 = (int*)take((size_t)16 * 1024 * NB_K * 4);
  int*   cnt1 = (int*)take(16 * 1024 * 4);
  f16*   x1   = (f16*)take((size_t)16 * 1024 * 128 * 2);
  int*   idx2 = (int*)take(16 * 256 * 4);
  float* c2   = (float*)take(16 * 256 * 3 * 4);
  int*   nbr2 = (int*)take((size_t)16 * 256 * NB_K * 4);
  int*   cnt2 = (int*)take(16 * 256 * 4);
  f16*   x2   = (f16*)take((size_t)16 * 256 * 256 * 2);
  float* part3 = (float*)take(16 * 4 * 1024 * 4);
  float* gbuf  = (float*)take(16 * 1024 * 4);

  LayerPtrs lp[9];
  for (int i = 0; i < 9; ++i) {
    f16*   wq = (f16*)take((size_t)KPAD[i] * COUT[i] * 2);
    float* bq = (float*)take(COUT[i] * 4);
    float* aq = (float*)take(COUT[i] * 4);
    float* sq = (float*)take(COUT[i] * 4);
    lp[i].w = wq; lp[i].b = bq; lp[i].a = aq; lp[i].s = sq;
    const int total = KPAD[i] * COUT[i];
    prep_layer_kernel<<<(total + 255) / 256, 256, 0, stream>>>(
        LIN(i / 3, i % 3, 0), LIN(i / 3, i % 3, 1), LIN(i / 3, i % 3, 2),
        LIN(i / 3, i % 3, 3), LIN(i / 3, i % 3, 4), LIN(i / 3, i % 3, 5),
        CIN[i], COUT[i], KPAD[i], wq, bq, aq, sq);
  }

  // SA1
  fps_kernel<2048, 1024, 8><<<16, 256, 0, stream>>>(POS, idx1, c1);
  ballq_kernel<<<(16 * 1024) / 256, 256, 0, stream>>>(POS, c1, nbr1, cnt1,
                                                      2048, 1024, 0.04f);
  mlp1_kernel<<<16 * 1024, 256, MLP1_LDS, stream>>>(POS, c1, nbr1, cnt1,
                                                    lp[0], lp[1], lp[2], x1);
  // SA2
  fps_kernel<1024, 256, 4><<<16, 256, 0, stream>>>(c1, idx2, c2);
  ballq_kernel<<<(16 * 256) / 256, 256, 0, stream>>>(c1, c2, nbr2, cnt2,
                                                     1024, 256, 0.16f);
  mlp2_kernel<<<16 * 256, 256, MLP2_LDS, stream>>>(x1, c1, c2, nbr2, cnt2,
                                                   lp[3], lp[4], lp[5], x2);
  // Global feature + head
  mlp3_kernel<<<16 * 4, 256, MLP3_LDS, stream>>>(x2, c2, lp[6], lp[7], lp[8], part3);
  gmax_kernel<<<(16 * 1024) / 256, 256, 0, stream>>>(part3, gbuf);
  head_kernel<<<16, 256, 0, stream>>>(gbuf, W1, B1, W2, B2, W3, B3, (float*)d_out);
}
